// GATv2Conv_20693152432941
// MI455X (gfx1250) — matrix operand order repaired and checked by reading, not compiled, not run on hardware
//
#include <hip/hip_runtime.h>
#include <hip/hip_bf16.h>
#include <stdint.h>

// Problem constants (from reference)
#define N_NODES 50000
#define N_EDGES 800000
#define DIM     128
#define HEADS   4
#define MTILES  (N_NODES / 16)   // 3125, exact

// LDS row stride for TDM-padded weight tiles: 64 data DWORDs + 4 pad DWORDs
// = 68 DWORDs = 272 B = 136 halves -> conflict-free ds_load_b128 per half-wave
#define WROW    136

typedef __attribute__((ext_vector_type(16))) __bf16 v16bf;
typedef __attribute__((ext_vector_type(8)))  float  v8f;
typedef unsigned v4u __attribute__((ext_vector_type(4)));
typedef int      v8i __attribute__((ext_vector_type(8)));
typedef int      v4i __attribute__((ext_vector_type(4)));

// ---- helpers -------------------------------------------------------------

__device__ __forceinline__ unsigned short f2bf(float f) {
  unsigned b = __float_as_uint(f);
  b += 0x7FFFu + ((b >> 16) & 1u);          // round-to-nearest-even
  return (unsigned short)(b >> 16);
}

// monotone float<->uint map for atomicMax on floats
__device__ __forceinline__ unsigned fkey(float f) {
  unsigned b = __float_as_uint(f);
  return (b & 0x80000000u) ? ~b : (b | 0x80000000u);
}
__device__ __forceinline__ float fdecode(unsigned k) {
  unsigned b = (k & 0x80000000u) ? (k ^ 0x80000000u) : ~k;
  return __uint_as_float(b);
}

// ---- Tensor Data Mover: DMA one 128x128 bf16 matrix into LDS -------------
// D# per CDNA5 ISA.
// Group0: count=1 | lds_addr | global_addr[31:0] | global_addr[56:32], type=2
// Group1: data_size=1 (2B), pad_enable, pad_interval=5 (64 DW), pad_amount=3
//         (4 DW); tensor_dim0=128, tensor_dim1=128, tile_dim0=128,
//         tile_dim1=128, tile_dim2=0, tensor_dim0_stride=128.
__device__ __forceinline__ void tdm_load_w(const unsigned short* gsrc,
                                           unsigned lds_off) {
  unsigned long long ga = (unsigned long long)(uintptr_t)gsrc;
  v4u g0 = { 1u,                                             // count=1, user D#
             lds_off,                                        // lds_addr (bytes)
             (unsigned)(ga & 0xFFFFFFFFu),                   // global_addr lo
             (unsigned)(((ga >> 32) & 0x01FFFFFFu) | (2u << 30)) }; // hi|type=2
  v8i g1 = { (int)(0x00010000u            // data_size = 1 (2-byte elements)
                   | (1u << 20)           // pad_enable
                   | (5u << 22)           // pad_interval code 5 = 64 DWORDs
                   | (3u << 25)),         // pad_amount   code 3 = 4 DWORDs
             (int)(128u << 16),           // tensor_dim0 = 128 (bits 79:48 lo)
             (int)(128u << 16),           // dim0 hi=0 | tensor_dim1 = 128 (lo)
             (int)(128u << 16),           // dim1 hi=0 | tile_dim0 = 128
             128,                         // tile_dim1 = 128, tile_dim2 = 0
             128,                         // tensor_dim0_stride = 128 (lo 32)
             0,                           // stride0 hi | tensor_dim1_stride lo
             0 };                         // tensor_dim1_stride hi
  v4i z = {0, 0, 0, 0};
#if __clang_major__ >= 23
  v8i z8 = {0, 0, 0, 0, 0, 0, 0, 0};
  __builtin_amdgcn_tensor_load_to_lds(g0, g1, z, z, z8, 0);
#else
  __builtin_amdgcn_tensor_load_to_lds(g0, g1, z, z, 0);
#endif
}

// ---- kernel 1: convert x -> bf16, zero accum/denom, reset global max -----

__global__ void k_prep(const float* __restrict__ x, unsigned short* __restrict__ xb,
                       float* __restrict__ accum, float* __restrict__ denom,
                       unsigned* __restrict__ gmax) {
  int i = blockIdx.x * blockDim.x + threadIdx.x;    // 6.4M threads == N*DIM
  xb[i]    = f2bf(x[i]);
  accum[i] = 0.f;
  if (i < N_NODES * HEADS) denom[i] = 0.f;
  if (i == 0) *gmax = 0u;                            // key 0 <= key(any finite float)
}

// ---- kernel 2: transpose-convert W[in][out] -> WT[out][in] bf16 ----------

__global__ void k_prep_w(const float* __restrict__ Ws, const float* __restrict__ Wd,
                         unsigned short* __restrict__ wtS, unsigned short* __restrict__ wtD) {
  int i = blockIdx.x * blockDim.x + threadIdx.x;    // 0 .. 32767
  int m = i & 16383;
  int n = m >> 7, k = m & 127;                       // WT[n*128+k] = W[k*128+n]
  if (i < 16384) wtS[m] = f2bf(Ws[k * DIM + n]);
  else           wtD[m] = f2bf(Wd[k * DIM + n]);
}

// ---- kernel 3: dual GEMM h_src = x@Wsrc, h_dst = x@Wdst ------------------
// Weights staged once per block via TDM (tensor_load_to_lds, TENSORcnt),
// B fragments from LDS (ds_load_b128, bank-conflict-free via TDM pad),
// A fragments resident in VGPRs. Both GEMMs computed together: alternating
// WMMAs on two independent accumulators cover the WMMA->WMMA RAW hazard
// window without v_nops and keep live registers ~80 (no spills).

__global__ void k_gemm(const unsigned short* __restrict__ xb,
                       const unsigned short* __restrict__ wtS,
                       const unsigned short* __restrict__ wtD,
                       float* __restrict__ hS, float* __restrict__ hD) {
  __shared__ unsigned short lds_w[2 * DIM * WROW + 8];   // ~68 KB + pad slack

  // One wave issues both DMAs (TDM ignores EXEC but branch-skipped waves
  // never issue); it drains TENSORcnt, then the block barrier publishes LDS.
  if (threadIdx.x < 32) {
    tdm_load_w(wtS, (unsigned)(uintptr_t)&lds_w[0]);
    tdm_load_w(wtD, (unsigned)(uintptr_t)&lds_w[DIM * WROW]);
    __builtin_amdgcn_s_wait_tensorcnt(0);
  }
  __syncthreads();

  const int mtile = blockIdx.x * 8 + (threadIdx.x >> 5);
  if (mtile >= MTILES) return;
  const int  lane = threadIdx.x & 31;
  const bool hi   = lane >= 16;
  const int  l15  = lane & 15;

  // A: 16x32 bf16 fragments: lane-lo halves0-15 K=kk..+15,
  // lane-hi +16. Two b128 loads each, kept resident.
  union AF { v16bf v; uint4 q[2]; } a[4];
  const unsigned short* arow = xb + (size_t)(mtile * 16 + l15) * DIM + (hi ? 16 : 0);
#pragma unroll
  for (int kk = 0; kk < 4; ++kk) {
    a[kk].q[0] = *(const uint4*)(arow + kk * 32);
    a[kk].q[1] = *(const uint4*)(arow + kk * 32 + 8);
  }

  // Per-lane base into the LDS weight tiles: row = column n (= lane&15),
  // +16 halves for lane-hi K offset (B: 32x16, lane-lo K=kk..+15, lane-hi +16).
  const unsigned short* b0base = &lds_w[0]          + (size_t)l15 * WROW + (hi ? 16 : 0);
  const unsigned short* b1base = &lds_w[DIM * WROW] + (size_t)l15 * WROW + (hi ? 16 : 0);

#pragma unroll 1
  for (int nt = 0; nt < 8; ++nt) {
    const unsigned short* br0 = b0base + (size_t)(nt * 16) * WROW;
    const unsigned short* br1 = b1base + (size_t)(nt * 16) * WROW;
    v8f acc0 = {0.f, 0.f, 0.f, 0.f, 0.f, 0.f, 0.f, 0.f};
    v8f acc1 = {0.f, 0.f, 0.f, 0.f, 0.f, 0.f, 0.f, 0.f};
#pragma unroll
    for (int kk = 0; kk < 4; ++kk) {
      union BF { v16bf v; uint4 q[2]; } b0, b1;
      b0.q[0] = *(const uint4*)(br0 + kk * 32);
      b0.q[1] = *(const uint4*)(br0 + kk * 32 + 8);
      b1.q[0] = *(const uint4*)(br1 + kk * 32);
      b1.q[1] = *(const uint4*)(br1 + kk * 32 + 8);
      acc0 = __builtin_amdgcn_wmma_f32_16x16x32_bf16(false, a[kk].v, false, b0.v,
                                                     (short)0, acc0, false, false);
      acc1 = __builtin_amdgcn_wmma_f32_16x16x32_bf16(false, a[kk].v, false, b1.v,
                                                     (short)0, acc1, false, false);
    }
    // C/D layout: VGPR r -> M = r (+8 for lane hi), N = lane&15
    size_t obase = (size_t)(mtile * 16 + (hi ? 8 : 0)) * DIM + nt * 16 + l15;
    float* o0 = hS + obase;
    float* o1 = hD + obase;
#pragma unroll
    for (int r = 0; r < 8; ++r) {
      o0[(size_t)r * DIM] = acc0[r];
      o1[(size_t)r * DIM] = acc1[r];
    }
  }
}

// ---- kernel 4: per-edge alpha = leakyReLU(h_s+h_d) @ W_attn, global max --

__global__ void k_alpha(const float* __restrict__ hS, const float* __restrict__ hD,
                        const int* __restrict__ ei, const float* __restrict__ Wattn,
                        float* __restrict__ alpha, unsigned* __restrict__ gmax) {
  __shared__ float wa[DIM * HEADS];
  __shared__ float red[256];
  for (int i = threadIdx.x; i < DIM * HEADS; i += blockDim.x) wa[i] = Wattn[i];
  __syncthreads();

  int e = blockIdx.x * blockDim.x + threadIdx.x;     // 3125*256 == E exactly
  int s = ei[e], d = ei[N_EDGES + e];
  const float4* ps = (const float4*)(hS + (size_t)s * DIM);
  const float4* pd = (const float4*)(hD + (size_t)d * DIM);
  float a0 = 0.f, a1 = 0.f, a2 = 0.f, a3 = 0.f;
#pragma unroll 4
  for (int c4 = 0; c4 < DIM / 4; ++c4) {
    float4 vs = ps[c4], vd = pd[c4];
    float v[4] = {vs.x + vd.x, vs.y + vd.y, vs.z + vd.z, vs.w + vd.w};
#pragma unroll
    for (int j = 0; j < 4; ++j) {
      float lv = v[j] > 0.f ? v[j] : 0.2f * v[j];    // leaky_relu(0.2)
      const float* w = &wa[(c4 * 4 + j) * HEADS];
      a0 += lv * w[0]; a1 += lv * w[1]; a2 += lv * w[2]; a3 += lv * w[3];
    }
  }
  *(float4*)(alpha + (size_t)e * HEADS) = make_float4(a0, a1, a2, a3);

  red[threadIdx.x] = fmaxf(fmaxf(a0, a1), fmaxf(a2, a3));
  __syncthreads();
  for (int off = 128; off > 0; off >>= 1) {
    if (threadIdx.x < off)
      red[threadIdx.x] = fmaxf(red[threadIdx.x], red[threadIdx.x + off]);
    __syncthreads();
  }
  if (threadIdx.x == 0) atomicMax(gmax, fkey(red[0]));
}

// ---- kernel 5: alpha_exp = exp(alpha - max); denom[dst] += alpha_exp -----

__global__ void k_exp(float* __restrict__ alpha, const int* __restrict__ ei,
                      float* __restrict__ denom, const unsigned* __restrict__ gmax) {
  int e = blockIdx.x * blockDim.x + threadIdx.x;
  float gm = fdecode(*gmax);
  int d = ei[N_EDGES + e];
  float4* pa = (float4*)(alpha + (size_t)e * HEADS);
  float4 a = *pa;
  a.x = __expf(a.x - gm); a.y = __expf(a.y - gm);
  a.z = __expf(a.z - gm); a.w = __expf(a.w - gm);
  *pa = a;                                           // in-place: now alpha_exp
  float* dn = denom + (size_t)d * HEADS;
  __hip_atomic_fetch_add(dn + 0, a.x, __ATOMIC_RELAXED, __HIP_MEMORY_SCOPE_AGENT);
  __hip_atomic_fetch_add(dn + 1, a.y, __ATOMIC_RELAXED, __HIP_MEMORY_SCOPE_AGENT);
  __hip_atomic_fetch_add(dn + 2, a.z, __ATOMIC_RELAXED, __HIP_MEMORY_SCOPE_AGENT);
  __hip_atomic_fetch_add(dn + 3, a.w, __ATOMIC_RELAXED, __HIP_MEMORY_SCOPE_AGENT);
}

// ---- kernel 6: one wave32 per edge: accum[dst] += h_src[src] * alpha_norm -

__global__ void k_msg(const float* __restrict__ hS, const float* __restrict__ alpha,
                      const float* __restrict__ denom, const int* __restrict__ ei,
                      float* __restrict__ accum) {
  int e    = blockIdx.x * 8 + (threadIdx.x >> 5);   // 100000 blocks * 8 waves == E
  int lane = threadIdx.x & 31;
  int s = ei[e], d = ei[N_EDGES + e];
  float4 a  = *(const float4*)(alpha + (size_t)e * HEADS);
  float4 dn = *(const float4*)(denom + (size_t)d * HEADS);
  float w[4] = {a.x / (dn.x + 1e-9f), a.y / (dn.y + 1e-9f),
                a.z / (dn.z + 1e-9f), a.w / (dn.w + 1e-9f)};
  const float* hs = hS + (size_t)s * DIM;
  float* ac = accum + (size_t)d * DIM;
#pragma unroll
  for (int h = 0; h < HEADS; ++h) {
    int c = lane + 32 * h;                           // head = c/32, coalesced
    __hip_atomic_fetch_add(ac + c, hs[c] * w[h],
                           __ATOMIC_RELAXED, __HIP_MEMORY_SCOPE_AGENT);
  }
}

// ---- kernel 7: out = LayerNorm(accum + h_dst) * gamma + beta -------------

__global__ void k_final(const float* __restrict__ accum, const float* __restrict__ hD,
                        const float* __restrict__ gamma, const float* __restrict__ beta,
                        float* __restrict__ out) {
  int n    = blockIdx.x * 8 + (threadIdx.x >> 5);   // 6250 blocks * 8 waves == N
  int lane = threadIdx.x & 31;
  int c    = lane * 4;
  float4 va = *(const float4*)(accum + (size_t)n * DIM + c);
  float4 vh = *(const float4*)(hD    + (size_t)n * DIM + c);
  float v[4] = {va.x + vh.x, va.y + vh.y, va.z + vh.z, va.w + vh.w};
  float s  = v[0] + v[1] + v[2] + v[3];
  float sq = v[0]*v[0] + v[1]*v[1] + v[2]*v[2] + v[3]*v[3];
#pragma unroll
  for (int m = 16; m > 0; m >>= 1) {                 // wave32 reduction
    s  += __shfl_xor(s,  m, 32);
    sq += __shfl_xor(sq, m, 32);
  }
  float mu  = s * (1.f / DIM);
  float var = sq * (1.f / DIM) - mu * mu;
  float inv = rsqrtf(var + 1e-5f);
  float4 g = *(const float4*)(gamma + c);
  float4 b = *(const float4*)(beta  + c);
  float* o = out + (size_t)n * DIM + c;
  o[0] = (v[0] - mu) * inv * g.x + b.x;
  o[1] = (v[1] - mu) * inv * g.y + b.y;
  o[2] = (v[2] - mu) * inv * g.z + b.z;
  o[3] = (v[3] - mu) * inv * g.w + b.w;
}

// ---- launcher ------------------------------------------------------------

extern "C" void kernel_launch(void* const* d_in, const int* in_sizes, int n_in,
                              void* d_out, int out_size, void* d_ws, size_t ws_size,
                              hipStream_t stream) {
  const float* x     = (const float*)d_in[0];
  const int*   ei    = (const int*)d_in[1];      // edge_index, [2*E] flat
  const float* Wsrc  = (const float*)d_in[2];
  const float* Wdst  = (const float*)d_in[3];
  const float* Wattn = (const float*)d_in[4];
  const float* gamma = (const float*)d_in[5];
  const float* beta  = (const float*)d_in[6];
  float* out = (float*)d_out;

  // Workspace layout (bytes, 256-aligned). ~90.5 MB total; x_bf16 aliases
  // alpha_exp (disjoint lifetimes: x_bf16 dead after k_gemm, alpha written after).
  char* ws = (char*)d_ws;
  float*          hS    = (float*)(ws + 0);              // 25,600,000
  float*          hD    = (float*)(ws + 25600000);       // 25,600,000
  float*          accum = (float*)(ws + 51200000);       // 25,600,000
  float*          denom = (float*)(ws + 76800000);       //    800,000
  float*          alpha = (float*)(ws + 77600000);       // 12,800,000
  unsigned short* xb    = (unsigned short*)(ws + 77600000); // aliases alpha
  unsigned short* wtS   = (unsigned short*)(ws + 90400000); // 32,768
  unsigned short* wtD   = (unsigned short*)(ws + 90432768); // 32,768
  unsigned*       gmax  = (unsigned*)(ws + 90465536);

  k_prep  <<<25000, 256, 0, stream>>>(x, xb, accum, denom, gmax);
  k_prep_w<<<  128, 256, 0, stream>>>(Wsrc, Wdst, wtS, wtD);
  k_gemm  <<<  391, 256, 0, stream>>>(xb, wtS, wtD, hS, hD);   // 3128 waves, 3125 tiles
  k_alpha <<< 3125, 256, 0, stream>>>(hS, hD, ei, Wattn, alpha, gmax);
  k_exp   <<< 3125, 256, 0, stream>>>(alpha, ei, denom, gmax);
  k_msg   <<<100000, 256, 0, stream>>>(hS, alpha, denom, ei, accum);
  k_final <<< 6250, 256, 0, stream>>>(accum, hD, gamma, beta, out);
}
